// CausalMultiHypothesisGraphTransformerLayer_3204045603778
// MI455X (gfx1250) — matrix operand order repaired and checked by reading, not compiled, not run on hardware
//
#include <hip/hip_runtime.h>

typedef __attribute__((ext_vector_type(16))) __bf16 v16bf;
typedef __attribute__((ext_vector_type(8)))  float  v8f;
typedef __attribute__((ext_vector_type(4)))  unsigned int v4u;
typedef __attribute__((ext_vector_type(8)))  int v8i;
typedef __attribute__((ext_vector_type(4)))  int v4i;

union FragU { unsigned int u[8]; v16bf v; };

constexpr int kB = 8;      // batch
constexpr int kC = 256;    // channels
constexpr int kN = 1024;   // nodes (32x32)
constexpr int kM = 8192;   // rows = kN * kB
constexpr float kEps = 1e-5f;

// ---------------------------------------------------------------------------
// Generic bf16 WMMA GEMM: out = A[M,K] @ W[K,Nc] (+bias) with fused epilogues.
// Bt is W pre-transposed to [Nc,K] row-major (bf16).
// Block = 8 waves; block tile 128(M) x 64(N); wave w owns rows m0..m0+16.
// The full 64xK B tile is staged once into LDS by the Tensor Data Mover
// (tensor_load_to_lds, TENSORcnt), with +4dword row padding so the 16
// lane-rows of each ds_load_b128 fragment land on distinct bank groups.
// A fragments are double-buffered in registers (issued one k-step ahead).
// Epilogues:
//  0: outF = acc+bias
//  1: outB = bf16(relu(acc+bias))
//  2: z-gate: outB = bf16(sigmoid(z)*aux), outB2 = bf16((1-sigmoid(z))*aux)
//  3: outF = acc+bias ; outB = bf16(same)
//  4: outB = bf16(acc+bias+aux+aux2)
//  6: outF = acc+bias+aux
// ---------------------------------------------------------------------------
__global__ __launch_bounds__(256) void gemm_bf16_wmma(
    const __bf16* __restrict__ A, const __bf16* __restrict__ Bt,
    const float* __restrict__ bias, int K, int Ncol, int epi,
    float* __restrict__ outF, __bf16* __restrict__ outB, __bf16* __restrict__ outB2,
    const float* __restrict__ aux, const float* __restrict__ aux2)
{
    __shared__ __bf16 ldsB[64 * 520];   // 64 rows x (Kmax=512 + 8 pad) bf16

    const int lane = threadIdx.x & 31;
    const int wave = threadIdx.x >> 5;
    const int m0 = (blockIdx.x * 8 + wave) * 16;
    const int n0 = blockIdx.y * 64;
    const int half = lane >> 4;     // 0: lanes 0-15, 1: lanes 16-31
    const int mr = lane & 15;
    const int rowStride = K + 8;    // bf16 elements per LDS row (+16B TDM pad)

    if (wave == 0) {
        // ---- Tensor DMA: rows [n0, n0+64) x [0, K) of Bt -> LDS ----
        const unsigned ldsAddr =
            (unsigned)(unsigned long long)(size_t)&ldsB[0];
        const unsigned long long ga =
            (unsigned long long)(size_t)(Bt + (size_t)n0 * K);
        const int padI = (31 - __builtin_clz((unsigned)K)) - 2;  // rows of K/2 dwords

        v4u g0;
        g0[0] = 1u;                                   // count=1, user mode
        g0[1] = ldsAddr;                              // lds_addr
        g0[2] = (unsigned)(ga & 0xffffffffull);       // global_addr lo
        g0[3] = (unsigned)(ga >> 32) | (2u << 30);    // global_addr hi | type=2

        v8i g1;
        g1[0] = (1 << 16)            // data_size = 2 bytes
              | (1 << 20)            // pad_enable
              | (padI << 22)         // pad_interval: one row of K elements
              | (3 << 25);           // pad_amount: 4 dwords (16B) per row
        g1[1] = (int)((unsigned)K << 16);     // tensor_dim0[15:0]
        g1[2] = (int)((unsigned)Ncol << 16);  // tensor_dim1[15:0]
        g1[3] = (int)((unsigned)K << 16);     // tile_dim0 = K
        g1[4] = 64;                           // tile_dim1 = 64 rows
        g1[5] = K;                            // tensor_dim0_stride
        g1[6] = 0;
        g1[7] = 0;

        v4i gz4 = {0, 0, 0, 0};               // groups 2/3: unused (2D tile)
        v8i gz8 = {0, 0, 0, 0, 0, 0, 0, 0};   // extra group (6-arg toolchain form)
        __builtin_amdgcn_tensor_load_to_lds(g0, g1, gz4, gz4, gz8, 0);
        __builtin_amdgcn_s_wait_tensorcnt(0);
    }
    __syncthreads();

    v8f acc[4] = {};

    const __bf16* Abase = A + (size_t)(m0 + mr) * K + 8 * half;
    FragU a_cur, a_nxt;
    #pragma unroll
    for (int v = 0; v < 8; ++v) {
        const int kk = (v < 4) ? (2 * v) : (16 + 2 * (v - 4));
        a_cur.u[v] = *(const unsigned int*)(Abase + kk);
    }

    const int nk = K >> 5;
    for (int i = 0; i < nk; ++i) {
        const int k0 = i << 5;
        if (i + 1 < nk) {           // issue next A fragment ahead of the WMMAs
            const __bf16* An = Abase + k0 + 32;
            #pragma unroll
            for (int v = 0; v < 8; ++v) {
                const int kk = (v < 4) ? (2 * v) : (16 + 2 * (v - 4));
                a_nxt.u[v] = *(const unsigned int*)(An + kk);
            }
        }
        #pragma unroll
        for (int s = 0; s < 4; ++s) {
            // B fragment: row (s*16+mr), k = k0+8*half+{0..7} and k0+16+8*half+{0..7}
            const v4u* bp_ = (const v4u*)(ldsB + (size_t)(s * 16 + mr) * rowStride
                                          + k0 + 8 * half);
            FragU b;
            v4u q0 = bp_[0], q1 = bp_[2];
            b.u[0] = q0[0]; b.u[1] = q0[1]; b.u[2] = q0[2]; b.u[3] = q0[3];
            b.u[4] = q1[0]; b.u[5] = q1[1]; b.u[6] = q1[2]; b.u[7] = q1[3];
            acc[s] = __builtin_amdgcn_wmma_f32_16x16x32_bf16(
                false, a_cur.v, false, b.v, (short)0, acc[s], false, false);
        }
        if (i + 1 < nk) a_cur = a_nxt;
    }

    // Epilogue (C/D layout: lane n = n0+s*16+(lane&15); VGPR r -> m = m0+r+8*half)
    #pragma unroll
    for (int s = 0; s < 4; ++s) {
        const int n = n0 + s * 16 + mr;
        const float bb = bias ? bias[n] : 0.f;
        #pragma unroll
        for (int r = 0; r < 8; ++r) {
            const int m = m0 + r + 8 * half;
            const size_t idx = (size_t)m * Ncol + n;
            const float val = acc[s][r] + bb;
            switch (epi) {
            case 0: outF[idx] = val; break;
            case 1: outB[idx] = (__bf16)(val > 0.f ? val : 0.f); break;
            case 2: {
                const float nd = aux[idx];
                const float sg = 1.f / (1.f + __expf(-val));
                outB[idx]  = (__bf16)(sg * nd);
                outB2[idx] = (__bf16)((1.f - sg) * nd);
            } break;
            case 3: outF[idx] = val; outB[idx] = (__bf16)val; break;
            case 4: outB[idx] = (__bf16)(val + aux[idx] + aux2[idx]); break;
            case 6: outF[idx] = val + aux[idx]; break;
            default: break;
            }
        }
    }
}

// ---------------------------------------------------------------------------
// Pack x[B,C,H,W] -> Xn[m=n*8+b, c] (bf16)
// ---------------------------------------------------------------------------
__global__ __launch_bounds__(256) void pack_x(const float* __restrict__ x,
                                              __bf16* __restrict__ Xn)
{
    const size_t i = (size_t)blockIdx.x * 256 + threadIdx.x;  // i = m*256 + c
    const int c = (int)(i & 255);
    const int m = (int)(i >> 8);
    const int n = m >> 3, b = m & 7;
    Xn[i] = (__bf16)x[((size_t)b * kC + c) * kN + n];
}

// ---------------------------------------------------------------------------
// Weight convert + transpose: Wt[n*K+k] = bf16(W[k*Nc+n])
// ---------------------------------------------------------------------------
__global__ __launch_bounds__(256) void wconv_t(const float* __restrict__ W,
                                               __bf16* __restrict__ Wt,
                                               int K, int Ncol)
{
    const int i = blockIdx.x * 256 + threadIdx.x;
    if (i >= K * Ncol) return;
    const int k = i / Ncol;
    const int n = i - k * Ncol;
    Wt[(size_t)n * K + k] = (__bf16)W[i];
}

// ---------------------------------------------------------------------------
// GCN aggregation (4-connected grid + self-loop, symmetric norm) + LayerNorm.
// One block per row m = n*8+b; 256 threads = channels.
// ---------------------------------------------------------------------------
__device__ __forceinline__ float grid_deg(int y, int x) {
    return 1.f + (x > 0) + (x < 31) + (y > 0) + (y < 31);
}

__global__ __launch_bounds__(256) void gcn_agg_ln(
    const float* __restrict__ hw, const float* __restrict__ bg,
    const float* __restrict__ g, const float* __restrict__ bl,
    float* __restrict__ outF, __bf16* __restrict__ outB)
{
    __shared__ float red[256];
    const int m = blockIdx.x;
    const int n = m >> 3, b = m & 7;
    const int c = threadIdx.x;
    const int y = n >> 5, xp = n & 31;

    int srcs[5]; float nrm[5]; int ne = 0;
    const float dd = grid_deg(y, xp);
    srcs[ne] = n;      nrm[ne] = 1.f / dd; ++ne;  // self loop: rsqrt(dd*dd)
    if (y > 0)   { srcs[ne] = n - 32; nrm[ne] = rsqrtf(grid_deg(y - 1, xp) * dd); ++ne; }
    if (y < 31)  { srcs[ne] = n + 32; nrm[ne] = rsqrtf(grid_deg(y + 1, xp) * dd); ++ne; }
    if (xp > 0)  { srcs[ne] = n - 1;  nrm[ne] = rsqrtf(grid_deg(y, xp - 1) * dd); ++ne; }
    if (xp < 31) { srcs[ne] = n + 1;  nrm[ne] = rsqrtf(grid_deg(y, xp + 1) * dd); ++ne; }

    float acc = bg ? bg[c] : 0.f;
    for (int e = 0; e < ne; ++e)
        acc += nrm[e] * hw[((size_t)(srcs[e] * kB + b)) * kC + c];

    // LayerNorm over C=256
    red[c] = acc; __syncthreads();
    for (int off = 128; off > 0; off >>= 1) { if (c < off) red[c] += red[c + off]; __syncthreads(); }
    const float mean = red[0] * (1.f / 256.f);
    __syncthreads();
    const float d = acc - mean;
    red[c] = d * d; __syncthreads();
    for (int off = 128; off > 0; off >>= 1) { if (c < off) red[c] += red[c + off]; __syncthreads(); }
    const float var = red[0] * (1.f / 256.f);
    const float o = d * rsqrtf(var + kEps) * g[c] + bl[c];
    outF[(size_t)m * kC + c] = o;
    if (outB) outB[(size_t)m * kC + c] = (__bf16)o;
}

// ---------------------------------------------------------------------------
// orig = adj + conf (f32 + bf16 mirror)
// ---------------------------------------------------------------------------
__global__ __launch_bounds__(256) void add_orig(const float* __restrict__ a,
                                                const float* __restrict__ b,
                                                float* __restrict__ of,
                                                __bf16* __restrict__ ob)
{
    const size_t i = (size_t)blockIdx.x * 256 + threadIdx.x;
    const float v = a[i] + b[i];
    of[i] = v; ob[i] = (__bf16)v;
}

// ---------------------------------------------------------------------------
// GATv2 gather + segment-softmax + head-mean. One block per (n,b); wave = head,
// lane covers 8 channel dims. Writes directly into d_out image layout.
// ---------------------------------------------------------------------------
__global__ __launch_bounds__(256) void gatv2_agg(
    const float* __restrict__ xl, const float* __restrict__ xr,
    const float* __restrict__ att, const float* __restrict__ gb,
    float* __restrict__ dout)
{
    __shared__ float smem[2048];
    const int m = blockIdx.x;
    const int n = m >> 3, b = m & 7;
    const int lane = threadIdx.x & 31;
    const int h = threadIdx.x >> 5;
    const int y = n >> 5, xp = n & 31;

    int srcs[5]; int ne = 0;
    srcs[ne++] = n;
    if (y > 0)   srcs[ne++] = n - 32;
    if (y < 31)  srcs[ne++] = n + 32;
    if (xp > 0)  srcs[ne++] = n - 1;
    if (xp < 31) srcs[ne++] = n + 1;

    const int d0 = lane * 8;
    const float* xrp = xr + (size_t)m * 2048 + h * 256 + d0;
    float xre[8], av[8];
    #pragma unroll
    for (int j = 0; j < 8; ++j) { xre[j] = xrp[j]; av[j] = att[h * 256 + d0 + j]; }

    float xle[5][8]; float logit[5];
    for (int e = 0; e < ne; ++e) {
        const float* xlp = xl + ((size_t)(srcs[e] * kB + b)) * 2048 + h * 256 + d0;
        float part = 0.f;
        #pragma unroll
        for (int j = 0; j < 8; ++j) {
            const float t = xlp[j];
            xle[e][j] = t;
            const float s = t + xre[j];
            part += av[j] * (s > 0.f ? s : 0.2f * s);   // leaky_relu(0.2)
        }
        #pragma unroll
        for (int off = 16; off >= 1; off >>= 1) part += __shfl_xor(part, off, 32);
        logit[e] = part;
    }

    float mx = logit[0];
    for (int e = 1; e < ne; ++e) mx = fmaxf(mx, logit[e]);
    float w[5]; float se = 0.f;
    for (int e = 0; e < ne; ++e) { w[e] = __expf(logit[e] - mx); se += w[e]; }
    const float inv = 1.f / se;

    float acc[8];
    #pragma unroll
    for (int j = 0; j < 8; ++j) acc[j] = 0.f;
    for (int e = 0; e < ne; ++e) {
        const float a = w[e] * inv;
        #pragma unroll
        for (int j = 0; j < 8; ++j) acc[j] += a * xle[e][j];
    }
    #pragma unroll
    for (int j = 0; j < 8; ++j) smem[h * 256 + d0 + j] = acc[j];
    __syncthreads();

    const int c = threadIdx.x;
    float o = 0.f;
    #pragma unroll
    for (int hh = 0; hh < 8; ++hh) o += smem[hh * 256 + c];
    o = o * 0.125f + gb[c];                      // head mean + bias
    dout[((size_t)b * kC + c) * kN + n] = o;
}

// ---------------------------------------------------------------------------
// Final LayerNorm of (orig+ffn) with transposed write into d_out.
// ---------------------------------------------------------------------------
__global__ __launch_bounds__(256) void ln_out(const float* __restrict__ t,
                                              const float* __restrict__ g,
                                              const float* __restrict__ bl,
                                              float* __restrict__ dout)
{
    __shared__ float red[256];
    const int m = blockIdx.x;
    const int n = m >> 3, b = m & 7;
    const int c = threadIdx.x;
    const float v = t[(size_t)m * kC + c];
    red[c] = v; __syncthreads();
    for (int off = 128; off > 0; off >>= 1) { if (c < off) red[c] += red[c + off]; __syncthreads(); }
    const float mean = red[0] * (1.f / 256.f);
    __syncthreads();
    const float d = v - mean;
    red[c] = d * d; __syncthreads();
    for (int off = 128; off > 0; off >>= 1) { if (c < off) red[c] += red[c + off]; __syncthreads(); }
    const float var = red[0] * (1.f / 256.f);
    const float o = d * rsqrtf(var + kEps) * g[c] + bl[c];
    dout[((size_t)b * kC + c) * kN + n] = o;
}

// ---------------------------------------------------------------------------
// Host orchestration
// ---------------------------------------------------------------------------
extern "C" void kernel_launch(void* const* d_in, const int* in_sizes, int n_in,
                              void* d_out, int out_size, void* d_ws, size_t ws_size,
                              hipStream_t stream)
{
    (void)in_sizes; (void)n_in; (void)out_size; (void)ws_size;
    const float* x    = (const float*)d_in[0];
    // d_in[1] = edge_index (int64) -- fixed 4-connected grid, recomputed on device
    const float* Wp   = (const float*)d_in[2];
    const float* bp   = (const float*)d_in[3];
    const float* Wm1  = (const float*)d_in[4];
    const float* bm1  = (const float*)d_in[5];
    const float* Wm2  = (const float*)d_in[6];
    const float* bm2  = (const float*)d_in[7];
    const float* Wgca = (const float*)d_in[8];
    const float* bgca = (const float*)d_in[9];
    const float* Wgcc = (const float*)d_in[10];
    const float* bgcc = (const float*)d_in[11];
    const float* Wi1  = (const float*)d_in[12];
    const float* bi1  = (const float*)d_in[13];
    const float* Wi2  = (const float*)d_in[14];
    const float* bi2  = (const float*)d_in[15];
    const float* Wl   = (const float*)d_in[16];
    const float* bl   = (const float*)d_in[17];
    const float* Wr   = (const float*)d_in[18];
    const float* br   = (const float*)d_in[19];
    const float* att  = (const float*)d_in[20];
    const float* gbias= (const float*)d_in[21];
    const float* Wf1  = (const float*)d_in[22];
    const float* bf1  = (const float*)d_in[23];
    const float* Wf2  = (const float*)d_in[24];
    const float* bf2  = (const float*)d_in[25];
    const float* lng  = (const float*)d_in[26];
    const float* lnb  = (const float*)d_in[27];
    float* out = (float*)d_out;

    // Workspace carve-out (deterministic layout, 256B aligned)
    char* p = (char*)d_ws;
    auto alloc = [&](size_t bytes) -> void* {
        void* r = (void*)p; p += (bytes + 255) & ~(size_t)255; return r;
    };
    __bf16* Xn    = (__bf16*)alloc((size_t)kM * kC * 2);
    float*  nodeF = (float*) alloc((size_t)kM * kC * 4);
    __bf16* nodeB = (__bf16*)alloc((size_t)kM * kC * 2);
    __bf16* hidB  = (__bf16*)alloc((size_t)kM * 128 * 2);
    __bf16* confB = (__bf16*)alloc((size_t)kM * kC * 2);
    __bf16* adjB  = (__bf16*)alloc((size_t)kM * kC * 2);
    float*  hwF   = (float*) alloc((size_t)kM * kC * 4);
    float*  adjF  = (float*) alloc((size_t)kM * kC * 4);
    float*  confF = (float*) alloc((size_t)kM * kC * 4);
    __bf16* cfeatB= (__bf16*)alloc((size_t)kM * kC * 2);
    __bf16* featB = (__bf16*)alloc((size_t)kM * kC * 2);
    float*  xlF   = (float*) alloc((size_t)kM * 2048 * 4);
    float*  xrF   = (float*) alloc((size_t)kM * 2048 * 4);
    float*  origF = (float*) alloc((size_t)kM * kC * 4);
    __bf16* origB = (__bf16*)alloc((size_t)kM * kC * 2);
    __bf16* ffnB  = (__bf16*)alloc((size_t)kM * 512 * 2);
    float*  tmpF  = (float*) alloc((size_t)kM * kC * 4);
    __bf16* WpT   = (__bf16*)alloc((size_t)65536 * 2);
    __bf16* Wm1T  = (__bf16*)alloc((size_t)32768 * 2);
    __bf16* Wm2T  = (__bf16*)alloc((size_t)32768 * 2);
    __bf16* WgcaT = (__bf16*)alloc((size_t)65536 * 2);
    __bf16* WgccT = (__bf16*)alloc((size_t)65536 * 2);
    __bf16* Wi1T  = (__bf16*)alloc((size_t)3 * 32768 * 2);
    __bf16* Wi2T  = (__bf16*)alloc((size_t)3 * 32768 * 2);
    __bf16* WlT   = (__bf16*)alloc((size_t)3 * 524288 * 2);
    __bf16* WrT   = (__bf16*)alloc((size_t)3 * 524288 * 2);
    __bf16* Wf1T  = (__bf16*)alloc((size_t)131072 * 2);
    __bf16* Wf2T  = (__bf16*)alloc((size_t)131072 * 2);

    auto wc = [&](const float* W, __bf16* Wt, int K, int Ncol) {
        const int nel = K * Ncol;
        wconv_t<<<dim3((nel + 255) / 256), dim3(256), 0, stream>>>(W, Wt, K, Ncol);
    };
    wc(Wp, WpT, 256, 256);
    wc(Wm1, Wm1T, 256, 128);
    wc(Wm2, Wm2T, 128, 256);
    wc(Wgca, WgcaT, 256, 256);
    wc(Wgcc, WgccT, 256, 256);
    for (int i = 0; i < 3; ++i) {
        wc(Wi1 + (size_t)i * 32768, Wi1T + (size_t)i * 32768, 256, 128);
        wc(Wi2 + (size_t)i * 32768, Wi2T + (size_t)i * 32768, 128, 256);
        wc(Wl + (size_t)i * 524288, WlT + (size_t)i * 524288, 256, 2048);
        wc(Wr + (size_t)i * 524288, WrT + (size_t)i * 524288, 256, 2048);
    }
    wc(Wf1, Wf1T, 256, 512);
    wc(Wf2, Wf2T, 512, 256);

    pack_x<<<dim3(kM), dim3(256), 0, stream>>>(x, Xn);

    auto gemm = [&](const __bf16* A, const __bf16* Bt, const float* bias,
                    int K, int Ncol, int epi, float* oF, __bf16* oB, __bf16* oB2,
                    const float* aux, const float* aux2) {
        gemm_bf16_wmma<<<dim3(kM / 128, Ncol / 64), dim3(256), 0, stream>>>(
            A, Bt, bias, K, Ncol, epi, oF, oB, oB2, aux, aux2);
    };

    // node = x@Wp + bp  (f32 + bf16)
    gemm(Xn, WpT, bp, 256, 256, 3, nodeF, nodeB, nullptr, nullptr, nullptr);
    // hid = relu(node@Wm1 + bm1)
    gemm(nodeB, Wm1T, bm1, 256, 128, 1, nullptr, hidB, nullptr, nullptr, nullptr);
    // z = hid@Wm2 + bm2 ; x_conf = sigmoid(z)*node ; x_adj = sigmoid(-z)*node
    gemm(hidB, Wm2T, bm2, 128, 256, 2, nullptr, confB, adjB, nodeF, nullptr);
    // GCN(adj): hw = x_adj@Wgca ; aggregate + bias + LN0
    gemm(adjB, WgcaT, nullptr, 256, 256, 0, hwF, nullptr, nullptr, nullptr, nullptr);
    gcn_agg_ln<<<dim3(kM), dim3(256), 0, stream>>>(hwF, bgca, lng + 0, lnb + 0,
                                                   adjF, (__bf16*)nullptr);
    // GCN(conf): hw = x_conf@Wgcc ; aggregate + bias + LN1 (bf16 mirror for interv GEMM)
    gemm(confB, WgccT, nullptr, 256, 256, 0, hwF, nullptr, nullptr, nullptr, nullptr);
    gcn_agg_ln<<<dim3(kM), dim3(256), 0, stream>>>(hwF, bgcc, lng + 256, lnb + 256,
                                                   confF, cfeatB);
    // orig = adj_feat + conf_feat
    add_orig<<<dim3(kM), dim3(256), 0, stream>>>(adjF, confF, origF, origB);

    for (int i = 0; i < 3; ++i) {
        // interv = relu(conf@Wi1+bi1)@Wi2+bi2 ; feat = adj+conf+interv (bf16)
        gemm(cfeatB, Wi1T + (size_t)i * 32768, bi1 + i * 128, 256, 128, 1,
             nullptr, hidB, nullptr, nullptr, nullptr);
        gemm(hidB, Wi2T + (size_t)i * 32768, bi2 + i * 256, 128, 256, 4,
             nullptr, featB, nullptr, adjF, confF);
        // xl / xr projections [8192, 2048]
        gemm(featB, WlT + (size_t)i * 524288, bl + (size_t)i * 2048, 256, 2048, 0,
             xlF, nullptr, nullptr, nullptr, nullptr);
        gemm(featB, WrT + (size_t)i * 524288, br + (size_t)i * 2048, 256, 2048, 0,
             xrF, nullptr, nullptr, nullptr, nullptr);
        // GATv2 attention + aggregation -> hyp i output image
        gatv2_agg<<<dim3(kM), dim3(256), 0, stream>>>(
            xlF, xrF, att + (size_t)i * 2048, gbias + (size_t)i * 256,
            out + (size_t)i * kB * kC * kN);
    }

    // FFN + residual + LN2 -> final output image
    gemm(origB, Wf1T, bf1, 256, 512, 1, nullptr, ffnB, nullptr, nullptr, nullptr);
    gemm(ffnB, Wf2T, bf2, 512, 256, 6, tmpF, nullptr, nullptr, origF, nullptr);
    ln_out<<<dim3(kM), dim3(256), 0, stream>>>(tmpF, lng + 512, lnb + 512,
                                               out + (size_t)3 * kB * kC * kN);
}
